// GraphAttentionLayer_30975304139274
// MI455X (gfx1250) — matrix operand order repaired and checked by reading, not compiled, not run on hardware
//
#include <hip/hip_runtime.h>
#include <hip/hip_fp16.h>

typedef __attribute__((ext_vector_type(16))) _Float16 v16h;
typedef __attribute__((ext_vector_type(8)))  float    v8f;
typedef __attribute__((ext_vector_type(4)))  float    v4f;

#define B_  8
#define N_  2048
#define F_  64
#define NEG_SLOPE 0.2f
#define MASK_VAL  -1000000000.0f

// ---------------------------------------------------------------------------
// Kernel A: h = X @ W  (per-row), store h^T as f16 [b][o][n] for WMMA-B frags,
// and compute ei = h . a_l, ej = h . a_r  via wave32 reductions.
// One 64-thread block per (b,n) row; thread o owns output feature o.
// ---------------------------------------------------------------------------
__global__ __launch_bounds__(64) void gat_proj_kernel(
    const float* __restrict__ x, const float* __restrict__ W,
    const float* __restrict__ a,
    _Float16* __restrict__ hT, float* __restrict__ ei, float* __restrict__ ej)
{
    const int row = blockIdx.x;          // 0 .. B*N-1
    const int b   = row / N_;
    const int n   = row % N_;
    const int o   = threadIdx.x;         // 0..63

    const float* xr = x + (size_t)row * F_;   // uniform per block -> scalar loads
    float acc = 0.f;
    #pragma unroll
    for (int f = 0; f < F_; ++f)
        acc = fmaf(xr[f], W[f * F_ + o], acc);

    hT[((size_t)b * F_ + o) * N_ + n] = (_Float16)acc;

    float pl = acc * a[o];
    float pr = acc * a[F_ + o];
    #pragma unroll
    for (int off = 16; off; off >>= 1) {   // wave32 tree reduce
        pl += __shfl_xor(pl, off, 32);
        pr += __shfl_xor(pr, off, 32);
    }
    __shared__ float sl[2], sr[2];
    const int wave = threadIdx.x >> 5;
    if ((threadIdx.x & 31) == 0) { sl[wave] = pl; sr[wave] = pr; }
    __syncthreads();
    if (threadIdx.x == 0) {
        ei[row] = sl[0] + sl[1];
        ej[row] = sr[0] + sr[1];
    }
}

// ---------------------------------------------------------------------------
// Kernel B: per-batch (max, 2nd max, argmax) of ej[b][:].  Because leaky_relu
// is monotone, row-max of scores is lrelu(ei[i] + (argmax==i ? max2 : max1)),
// eliminating online-softmax rescaling in the main kernel.
// ---------------------------------------------------------------------------
__global__ __launch_bounds__(256) void gat_stats_kernel(
    const float* __restrict__ ej, float* __restrict__ stats)
{
    const int b = blockIdx.x;
    const float* e = ej + (size_t)b * N_;
    float m1 = -INFINITY, m2 = -INFINITY; int a1 = -1;
    for (int n = threadIdx.x; n < N_; n += 256) {
        float v = e[n];
        if (v > m1)      { m2 = m1; m1 = v; a1 = n; }
        else if (v > m2) { m2 = v; }
    }
    __shared__ float sm1[256], sm2[256];
    __shared__ int   sa1[256];
    const int t = threadIdx.x;
    sm1[t] = m1; sm2[t] = m2; sa1[t] = a1;
    __syncthreads();
    for (int s = 128; s; s >>= 1) {
        if (t < s) {
            float o1 = sm1[t + s], o2 = sm2[t + s]; int oa = sa1[t + s];
            if (o1 > sm1[t]) { sm2[t] = fmaxf(sm1[t], o2); sm1[t] = o1; sa1[t] = oa; }
            else             { sm2[t] = fmaxf(sm2[t], o1); }
        }
        __syncthreads();
    }
    if (t == 0) {
        stats[b * 4 + 0] = sm1[0];
        stats[b * 4 + 1] = sm2[0];
        stats[b * 4 + 2] = __int_as_float(sa1[0]);
    }
}

// ---------------------------------------------------------------------------
// Kernel C: fused attention.  Each wave32 owns a 16-row i-tile; loops over j
// in tiles of 32: builds P = exp(lrelu(ei+ej) - mrow) in registers laid out
// as the WMMA f16 A-fragment (fully branchless: diag masked in the score, exp
// taken unconditionally; ej read as contiguous ds_load_b128 runs), loads h^T
// B-fragments (contiguous 32B per lane), and accumulates 4x
// v_wmma_f32_16x16x32_f16 (o-chunks of 16).  Row sums via lane^16 shuffle;
// final scale uses the C/D VGPR layout (VGPR r <-> rows r, r+8).
// ---------------------------------------------------------------------------
__global__ __launch_bounds__(128) void gat_attn_kernel(
    const _Float16* __restrict__ hT,
    const float* __restrict__ ei_g, const float* __restrict__ ej_g,
    const float* __restrict__ stats, float* __restrict__ out)
{
    const int BLOCKS_PER_B = N_ / 64;               // 32
    const int b    = blockIdx.x / BLOCKS_PER_B;
    const int rb   = blockIdx.x % BLOCKS_PER_B;
    const int wave = threadIdx.x >> 5;
    const int lane = threadIdx.x & 31;
    const int Mrow = lane & 15;                     // row within i-tile / B column
    const int khi  = lane >> 4;                     // lane half selects K subset
    const int ibase = rb * 64 + wave * 16;

    __shared__ float s_ej[N_];
    __shared__ float s_l[64];
    for (int n = threadIdx.x; n < N_; n += 128)
        s_ej[n] = ej_g[(size_t)b * N_ + n];
    __syncthreads();

    const int gi    = ibase + Mrow;
    const float eiv = ei_g[(size_t)b * N_ + gi];
    const float m1  = stats[b * 4 + 0];
    const float m2  = stats[b * 4 + 1];
    const int   a1  = __float_as_int(stats[b * 4 + 2]);
    float mx = eiv + ((a1 == gi) ? m2 : m1);        // closed-form row max
    const float mrow = fmaxf(mx, NEG_SLOPE * mx);   // lrelu (slope<1 => max form)

    v8f acc0 = {}, acc1 = {}, acc2 = {}, acc3 = {};
    float ps0 = 0.f, ps1 = 0.f, ps2 = 0.f, ps3 = 0.f;

    const _Float16* hTb = hT + (size_t)b * F_ * N_;

    for (int jbase = 0; jbase < N_; jbase += 32) {
        // ---- ej for this lane's 16 K-slots: two contiguous 8-float runs ----
        // e=0..7  -> j = jbase + khi*16 + e        (K = e + 16*khi)
        // e=8..15 -> j = jbase + khi*16 + 8 + e-8  (K = e + 16*khi)
        const v4f* r0 = (const v4f*)&s_ej[jbase + khi * 16];       // 2x ds_load_b128
        const v4f* r1 = (const v4f*)&s_ej[jbase + khi * 16 + 8];   // 2x ds_load_b128
        v4f e0 = r0[0], e1 = r0[1], e2 = r1[0], e3 = r1[1];
        float ev[16];
        #pragma unroll
        for (int q = 0; q < 4; ++q) {
            ev[q]      = e0[q];
            ev[4 + q]  = e1[q];
            ev[8 + q]  = e2[q];
            ev[12 + q] = e3[q];
        }

        // ---- build A fragment: fully branchless, unconditional v_exp_f32 ----
        v16h afrag;
        #pragma unroll
        for (int e = 0; e < 16; ++e) {
            const int K = e + 16 * khi;
            const int j = jbase + K;
            float s = eiv + ev[e];
            s = fmaxf(s, NEG_SLOPE * s);            // leaky_relu
            s = (j == gi) ? MASK_VAL : s;           // diag mask -> exp underflows to 0
            float p = __expf(s - mrow);
            if      ((e & 3) == 0) ps0 += p;        // 4-way partial sums: short dep chains
            else if ((e & 3) == 1) ps1 += p;
            else if ((e & 3) == 2) ps2 += p;
            else                   ps3 += p;
            afrag[e] = (_Float16)p;
        }

        // ---- B fragments: hT rows are features; lane reads 16 contiguous halves ----
        const _Float16* bbase = hTb + (size_t)Mrow * N_ + jbase + khi * 16;
        v16h b0 = *(const v16h*)(bbase);
        v16h b1 = *(const v16h*)(bbase + (size_t)16 * N_);
        v16h b2 = *(const v16h*)(bbase + (size_t)32 * N_);
        v16h b3 = *(const v16h*)(bbase + (size_t)48 * N_);

        acc0 = __builtin_amdgcn_wmma_f32_16x16x32_f16(false, afrag, false, b0, (short)0, acc0, false, false);
        acc1 = __builtin_amdgcn_wmma_f32_16x16x32_f16(false, afrag, false, b1, (short)0, acc1, false, false);
        acc2 = __builtin_amdgcn_wmma_f32_16x16x32_f16(false, afrag, false, b2, (short)0, acc2, false, false);
        acc3 = __builtin_amdgcn_wmma_f32_16x16x32_f16(false, afrag, false, b3, (short)0, acc3, false, false);
    }

    // row sum: lane L holds half the K range, lane L^16 the other half
    float psum = (ps0 + ps1) + (ps2 + ps3);
    psum += __shfl_xor(psum, 16, 32);
    s_l[wave * 16 + Mrow] = psum;          // lanes L and L+16 write identical value
    __syncthreads();

    // scale by 1/rowsum and store; C/D layout: VGPR r -> rows r (lanes0-15), r+8 (lanes16-31)
    #pragma unroll
    for (int r = 0; r < 8; ++r) {
        const int rowl = r + khi * 8;
        const float inv = 1.0f / s_l[wave * 16 + rowl];
        float* op = out + ((size_t)b * N_ + (ibase + rowl)) * F_ + Mrow;
        op[0]  = acc0[r] * inv;
        op[16] = acc1[r] * inv;
        op[32] = acc2[r] * inv;
        op[48] = acc3[r] * inv;
    }
}

// ---------------------------------------------------------------------------
extern "C" void kernel_launch(void* const* d_in, const int* in_sizes, int n_in,
                              void* d_out, int out_size, void* d_ws, size_t ws_size,
                              hipStream_t stream)
{
    const float* x = (const float*)d_in[0];   // node_features (B,N,64)
    // d_in[1] = edge_features: unused by the reference computation
    const float* W = (const float*)d_in[2];   // (64,64)
    const float* a = (const float*)d_in[3];   // (128,)
    float* out = (float*)d_out;               // (B,N,64)

    char* ws = (char*)d_ws;
    _Float16* hT  = (_Float16*)ws;                                   // 8*64*2048*2 = 2 MB
    float* ei     = (float*)(ws + (size_t)2 * 1024 * 1024);          // 64 KB
    float* ej     = (float*)(ws + (size_t)2 * 1024 * 1024 + 65536);  // 64 KB
    float* stats  = (float*)(ws + (size_t)2 * 1024 * 1024 + 131072); // 128 B

    gat_proj_kernel <<<B_ * N_,        64,  0, stream>>>(x, W, a, hT, ei, ej);
    gat_stats_kernel<<<B_,             256, 0, stream>>>(ej, stats);
    gat_attn_kernel <<<B_ * (N_ / 64), 128, 0, stream>>>(hT, ei, ej, stats, out);
}
